// TemplatePointwiseAttention_30391188586650
// MI455X (gfx1250) — matrix operand order repaired and checked by reading, not compiled, not run on hardware
//
#include <hip/hip_runtime.h>
#include <math.h>

// ---------------- problem constants ----------------
#define HEAD_DIM  64
#define NUM_HEADS 4
#define C_Z       128
#define TOTAL_DIM 256          // NUM_HEADS * HEAD_DIM
#define NPOS      65536        // 256 * 256 (i,j) positions
#define NTPL      4            // templates
#define TILE_M    64           // positions per workgroup
#define BLOCK     256          // 8 wave32

typedef __attribute__((ext_vector_type(16))) __bf16 bf16x16;
typedef __attribute__((ext_vector_type(8)))  float  f32x8;

union FragBF { bf16x16 v; uint4 q[2]; };

// ---------------- bf16 helpers (bit-exact RNE, no reliance on scalar __bf16 conv) ---
__device__ __forceinline__ unsigned short f2bf(float f) {
  union { float f; unsigned u; } c; c.f = f;
  unsigned u = c.u + 0x7fffu + ((c.u >> 16) & 1u);
  return (unsigned short)(u >> 16);
}
__device__ __forceinline__ float bflo(unsigned u) {
  union { unsigned u; float f; } c; c.u = u << 16; return c.f;
}
__device__ __forceinline__ float bfhi(unsigned u) {
  union { unsigned u; float f; } c; c.u = u & 0xffff0000u; return c.f;
}
__device__ __forceinline__ f32x8 zero8() {
  f32x8 z;
#pragma unroll
  for (int i = 0; i < 8; ++i) z[i] = 0.f;
  return z;
}

// ---------------- fragment loaders (bf16, row-major source) ----------------
// A 16x32 bf16 (ISA 7.12.2): lane m=l&15; half kb=(l>>4)*8; elems 0..7 = K kb..kb+7,
// elems 8..15 = K kb+16..kb+23. Both halves are 16B-contiguous -> 2x b128 loads.
__device__ __forceinline__ bf16x16 frag_a(const unsigned short* base, int row0, int k0,
                                          int stride) {
  const int lane = threadIdx.x & 31;
  const unsigned short* p =
      base + (row0 + (lane & 15)) * stride + k0 + ((lane >> 4) << 3);
  FragBF f;
  f.q[0] = *(const uint4*)(p);
  f.q[1] = *(const uint4*)(p + 16);
  return f.v;
}

// B 32x16 bf16 from weights stored [N][K] (torch Linear weight): B(k,n)=W[n][k].
// lane n=l&15; half kb=(l>>4)*8; elems 0..7 = K kb..kb+7, elems 8..15 = K kb+16..kb+23 -> 2x b128 loads.
__device__ __forceinline__ bf16x16 frag_b(const unsigned short* w, int n0, int k0,
                                          int K) {
  const int lane = threadIdx.x & 31;
  const unsigned short* p = w + (n0 + (lane & 15)) * K + k0 + ((lane >> 4) << 3);
  FragBF f;
  f.q[0] = *(const uint4*)(p);
  f.q[1] = *(const uint4*)(p + 16);
  return f.v;
}

// ---------------- cooperative fp32 -> bf16 tile staging (64 x 128) ----------------
__device__ __forceinline__ void stage_tile(unsigned short* dst, const float* src) {
  const int tid = threadIdx.x;
#pragma unroll
  for (int it = 0; it < 8; ++it) {
    int idx = it * BLOCK + tid;        // float4 index, 0..2047
    int row = idx >> 5;                // 32 float4 per 128-wide row
    int c   = (idx & 31) << 2;
    const float4 v = *(const float4*)(src + row * C_Z + c);
    uint2 p;
    p.x = (unsigned)f2bf(v.x) | ((unsigned)f2bf(v.y) << 16);
    p.y = (unsigned)f2bf(v.z) | ((unsigned)f2bf(v.w) << 16);
    *(uint2*)(dst + row * C_Z + c) = p;
  }
}

// ---------------- weight prep: fp32 -> bf16 into workspace ----------------
__global__ void prep_weights(const float* __restrict__ wq, const float* __restrict__ wk,
                             const float* __restrict__ wv, const float* __restrict__ wo,
                             unsigned short* __restrict__ dst) {
  int idx = blockIdx.x * blockDim.x + threadIdx.x;   // 0 .. 131071
  if (idx >= 4 * TOTAL_DIM * C_Z) return;
  float v;
  if (idx < 32768)       v = wq[idx];
  else if (idx < 65536)  v = wk[idx - 32768];
  else if (idx < 98304)  v = wv[idx - 65536];
  else                   v = wo[idx - 98304];
  dst[idx] = f2bf(v);
}

// ---------------- fused template pointwise attention ----------------
__global__ __launch_bounds__(BLOCK)
void tpa_kernel(const float* __restrict__ tfeat, const float* __restrict__ pair,
                const float* __restrict__ tmask,
                const unsigned short* __restrict__ wq_bf,
                const unsigned short* __restrict__ wk_bf,
                const unsigned short* __restrict__ wv_bf,
                const unsigned short* __restrict__ wo_bf,
                const float* __restrict__ bo, float* __restrict__ out) {
  extern __shared__ unsigned char smem[];
  unsigned short* xt = (unsigned short*)smem;           // 4 x 64 x 128  (64 KB)
  unsigned short* xp = xt + NTPL * TILE_M * C_Z;        // 64 x 128      (16 KB)
  unsigned short* qb = xp + TILE_M * C_Z;               // 64 x 256      (32 KB)
  unsigned short* kv = qb + TILE_M * TOTAL_DIM;         // 64 x 256      (32 KB)
  float* sc = (float*)(kv + TILE_M * TOTAL_DIM);        // 64 x 4h x 4t  ( 4 KB)

  const int pos0 = blockIdx.x * TILE_M;
  const int tid  = threadIdx.x;
  const int lane = tid & 31;
  const int wave = tid >> 5;
  const int row0 = (wave & 3) * 16;      // 16-row block owned by this wave
  const int nh   = wave >> 1 >> 1;       // 0/1: which half of the 256-wide N dim
  const int mh   = (lane >> 4) << 3;     // C/D row offset (ISA layout)
  const int ncol = lane & 15;            // C/D column (ISA layout)

  // ---- stage pair tile; Q = pair @ Wq^T ----
  stage_tile(xp, pair + (size_t)pos0 * C_Z);
  __syncthreads();
  {
    bf16x16 a[4];
#pragma unroll
    for (int kk = 0; kk < 4; ++kk) a[kk] = frag_a(xp, row0, kk * 32, C_Z);
#pragma unroll
    for (int nt = 0; nt < 8; ++nt) {
      const int n0 = nh * 128 + nt * 16;
      f32x8 acc = zero8();
#pragma unroll
      for (int kk = 0; kk < 4; ++kk)
        acc = __builtin_amdgcn_wmma_f32_16x16x32_bf16(
            false, a[kk], false, frag_b(wq_bf, n0, kk * 32, C_Z), (short)0, acc,
            false, false);
#pragma unroll
      for (int r = 0; r < 8; ++r)
        qb[(row0 + mh + r) * TOTAL_DIM + n0 + ncol] = f2bf(acc[r]);
    }
  }

  // ---- stage all 4 template tiles (reused for K and V projections) ----
  for (int t = 0; t < NTPL; ++t)
    stage_tile(xt + t * TILE_M * C_Z, tfeat + ((size_t)t * NPOS + pos0) * C_Z);
  __syncthreads();

  // ---- per-template: K projection + masked scores ----
  for (int t = 0; t < NTPL; ++t) {
    {
      bf16x16 a[4];
#pragma unroll
      for (int kk = 0; kk < 4; ++kk)
        a[kk] = frag_a(xt + t * TILE_M * C_Z, row0, kk * 32, C_Z);
#pragma unroll
      for (int nt = 0; nt < 8; ++nt) {
        const int n0 = nh * 128 + nt * 16;
        f32x8 acc = zero8();
#pragma unroll
        for (int kk = 0; kk < 4; ++kk)
          acc = __builtin_amdgcn_wmma_f32_16x16x32_bf16(
              false, a[kk], false, frag_b(wk_bf, n0, kk * 32, C_Z), (short)0,
              acc, false, false);
#pragma unroll
        for (int r = 0; r < 8; ++r)
          kv[(row0 + mh + r) * TOTAL_DIM + n0 + ncol] = f2bf(acc[r]);
      }
    }
    __syncthreads();
    {  // one thread per (position m, head h): dot over d=64
      const int m = tid >> 2, h = tid & 3;
      const unsigned short* qp = qb + m * TOTAL_DIM + h * HEAD_DIM;
      const unsigned short* kp = kv + m * TOTAL_DIM + h * HEAD_DIM;
      float s = 0.f;
#pragma unroll
      for (int d = 0; d < HEAD_DIM; d += 8) {
        uint4 qu = *(const uint4*)(qp + d);
        uint4 ku = *(const uint4*)(kp + d);
        s += bflo(qu.x) * bflo(ku.x) + bfhi(qu.x) * bfhi(ku.x);
        s += bflo(qu.y) * bflo(ku.y) + bfhi(qu.y) * bfhi(ku.y);
        s += bflo(qu.z) * bflo(ku.z) + bfhi(qu.z) * bfhi(ku.z);
        s += bflo(qu.w) * bflo(ku.w) + bfhi(qu.w) * bfhi(ku.w);
      }
      const float mk = tmask[t];
      sc[m * 16 + h * 4 + t] = s * 0.125f + (1.f - mk) * -1e9f;  // 1/sqrt(64)
    }
    __syncthreads();
  }

  // ---- softmax over t (4 values per (m,h)) ----
  {
    const int m = tid >> 2, h = tid & 3;
    float* sp = sc + m * 16 + h * 4;
    float s0 = sp[0], s1 = sp[1], s2 = sp[2], s3 = sp[3];
    float mx = fmaxf(fmaxf(s0, s1), fmaxf(s2, s3));
    float e0 = expf(s0 - mx), e1 = expf(s1 - mx);
    float e2 = expf(s2 - mx), e3 = expf(s3 - mx);
    float inv = 1.f / (e0 + e1 + e2 + e3);
    sp[0] = e0 * inv; sp[1] = e1 * inv; sp[2] = e2 * inv; sp[3] = e3 * inv;
  }
  __syncthreads();

  // ---- V projection + attention-weighted accumulation (in registers) ----
  f32x8 vac[8];
#pragma unroll
  for (int nt = 0; nt < 8; ++nt) vac[nt] = zero8();
  for (int t = 0; t < NTPL; ++t) {
    bf16x16 a[4];
#pragma unroll
    for (int kk = 0; kk < 4; ++kk)
      a[kk] = frag_a(xt + t * TILE_M * C_Z, row0, kk * 32, C_Z);
#pragma unroll
    for (int nt = 0; nt < 8; ++nt) {
      const int n0 = nh * 128 + nt * 16;
      const int h  = n0 >> 6;
      f32x8 acc = zero8();
#pragma unroll
      for (int kk = 0; kk < 4; ++kk)
        acc = __builtin_amdgcn_wmma_f32_16x16x32_bf16(
            false, a[kk], false, frag_b(wv_bf, n0, kk * 32, C_Z), (short)0, acc,
            false, false);
#pragma unroll
      for (int r = 0; r < 8; ++r) {
        const float w = sc[(row0 + mh + r) * 16 + h * 4 + t];
        vac[nt][r] += w * acc[r];
      }
    }
  }

  // restage weighted values as bf16 (reuse K buffer)
#pragma unroll
  for (int nt = 0; nt < 8; ++nt) {
    const int n0 = nh * 128 + nt * 16;
#pragma unroll
    for (int r = 0; r < 8; ++r)
      kv[(row0 + mh + r) * TOTAL_DIM + n0 + ncol] = f2bf(vac[nt][r]);
  }
  __syncthreads();

  // ---- out = values @ Wo^T + bo  (K=256, N=128) ----
  {
    bf16x16 a[8];
#pragma unroll
    for (int kk = 0; kk < 8; ++kk) a[kk] = frag_a(kv, row0, kk * 32, TOTAL_DIM);
#pragma unroll
    for (int nt = 0; nt < 4; ++nt) {
      const int n0 = nh * 64 + nt * 16;
      f32x8 acc = zero8();
#pragma unroll
      for (int kk = 0; kk < 8; ++kk)
        acc = __builtin_amdgcn_wmma_f32_16x16x32_bf16(
            false, a[kk], false, frag_b(wo_bf, n0, kk * 32, TOTAL_DIM), (short)0,
            acc, false, false);
      const float bias = bo[n0 + ncol];
#pragma unroll
      for (int r = 0; r < 8; ++r)
        out[(size_t)(pos0 + row0 + mh + r) * C_Z + n0 + ncol] = acc[r] + bias;
    }
  }
}

// ---------------- host launcher ----------------
extern "C" void kernel_launch(void* const* d_in, const int* in_sizes, int n_in,
                              void* d_out, int out_size, void* d_ws, size_t ws_size,
                              hipStream_t stream) {
  (void)in_sizes; (void)n_in; (void)out_size; (void)ws_size;
  const float* tfeat = (const float*)d_in[0];
  const float* pair  = (const float*)d_in[1];
  const float* tmask = (const float*)d_in[2];
  const float* Wq    = (const float*)d_in[3];
  const float* Wk    = (const float*)d_in[4];
  const float* Wv    = (const float*)d_in[5];
  const float* Wo    = (const float*)d_in[6];
  const float* bo    = (const float*)d_in[7];
  float* out = (float*)d_out;
  unsigned short* wbf = (unsigned short*)d_ws;   // 4 x 32768 bf16 = 256 KB

  prep_weights<<<(4 * TOTAL_DIM * C_Z + BLOCK - 1) / BLOCK, BLOCK, 0, stream>>>(
      Wq, Wk, Wv, Wo, wbf);

  const size_t smem =
      (size_t)(NTPL * TILE_M * C_Z + TILE_M * C_Z + 2 * TILE_M * TOTAL_DIM) *
          sizeof(unsigned short) +
      (size_t)TILE_M * 16 * sizeof(float);   // 151,552 B -> 2 WGs / 320KB WGP
  hipFuncSetAttribute((const void*)tpa_kernel,
                      hipFuncAttributeMaxDynamicSharedMemorySize, (int)smem);

  tpa_kernel<<<NPOS / TILE_M, BLOCK, smem, stream>>>(
      tfeat, pair, tmask, wbf, wbf + 32768, wbf + 65536, wbf + 98304, bo, out);
}
